// Layer1MeanAggregator_9603546873885
// MI455X (gfx1250) — matrix operand order repaired and checked by reading, not compiled, not run on hardware
//
#include <hip/hip_runtime.h>

#define D       128
#define S       25
#define TILE_M  16
#define WAVES   8
#define BLOCK   (WAVES * 32)
#define KDIM    (2 * D)        // 256
#define KPAD    264            // 256 + 8 halves padding (16B) -> conflict-free LDS
#define NODES_PER_BLOCK (WAVES * TILE_M)

typedef __attribute__((ext_vector_type(16))) _Float16 v16h;
typedef __attribute__((ext_vector_type(8)))  _Float16 v8h;
typedef __attribute__((ext_vector_type(4)))  _Float16 v4h;
typedef __attribute__((ext_vector_type(8)))  float    v8f;
typedef __attribute__((ext_vector_type(4)))  float    vf4;

union AFrag { v16h v; v8h h[2]; };

__global__ __launch_bounds__(BLOCK)
void sage_layer_fused(const float* __restrict__ self_rows,   // [N][128]
                      const float* __restrict__ neigh_rows,  // [N*25][128]
                      const float* __restrict__ w,           // [256][128] k-major
                      float* __restrict__ out,               // [N][128]
                      int N)
{
    __shared__ __align__(16) _Float16 sW[D * KPAD];                 // sW[n*KPAD+k] = w[k][n]
    __shared__ __align__(16) _Float16 sA[WAVES * TILE_M * KPAD];    // per-wave A tiles

    const int tid  = threadIdx.x;
    const int lane = tid & 31;
    const int wv   = tid >> 5;

    // ---------- Phase 0: transpose + f32->f16 convert weight into LDS ----------
    // w is small (128 KB) and shared by every block: keep it temporal (default TH)
    for (int idx = tid; idx < KDIM * D; idx += BLOCK) {
        const int k = idx >> 7;        // 0..255 (row of w)
        const int n = idx & (D - 1);   // 0..127 (col of w)
        sW[n * KPAD + k] = (_Float16)w[idx];   // coalesced global read
    }

    // ---------- Phase 1: mean-pool neighbors + build [self | agg] f16 tiles ----------
    // Streamed exactly once -> non-temporal loads; two nodes in flight for MLP.
    const int tileBase = (blockIdx.x * WAVES + wv) * TILE_M;
    {
        const float inv = 1.0f / (float)S;
        for (int i = 0; i < TILE_M; i += 2) {
            int n0 = tileBase + i;
            int n1 = tileBase + i + 1;
            n0 = n0 < N ? n0 : (N - 1);
            n1 = n1 < N ? n1 : (N - 1);

            const vf4* sp0 = (const vf4*)(self_rows + (size_t)n0 * D) + lane;
            const vf4* sp1 = (const vf4*)(self_rows + (size_t)n1 * D) + lane;
            const vf4 sv0 = __builtin_nontemporal_load(sp0);
            const vf4 sv1 = __builtin_nontemporal_load(sp1);

            const vf4* p0 = (const vf4*)(neigh_rows + (size_t)n0 * S * D) + lane;
            const vf4* p1 = (const vf4*)(neigh_rows + (size_t)n1 * S * D) + lane;
            vf4 a0 = {0.f, 0.f, 0.f, 0.f};
            vf4 a1 = {0.f, 0.f, 0.f, 0.f};
            #pragma unroll
            for (int s = 0; s < S; ++s) {
                const vf4 u = __builtin_nontemporal_load(p0 + s * (D / 4));
                const vf4 v = __builtin_nontemporal_load(p1 + s * (D / 4));
                a0 += u;
                a1 += v;
            }

            _Float16* row0 = &sA[(wv * TILE_M + i) * KPAD];
            _Float16* row1 = row0 + KPAD;
            v4h hs0 = { (_Float16)sv0.x, (_Float16)sv0.y, (_Float16)sv0.z, (_Float16)sv0.w };
            v4h hs1 = { (_Float16)sv1.x, (_Float16)sv1.y, (_Float16)sv1.z, (_Float16)sv1.w };
            v4h ha0 = { (_Float16)(a0.x * inv), (_Float16)(a0.y * inv),
                        (_Float16)(a0.z * inv), (_Float16)(a0.w * inv) };
            v4h ha1 = { (_Float16)(a1.x * inv), (_Float16)(a1.y * inv),
                        (_Float16)(a1.z * inv), (_Float16)(a1.w * inv) };
            ((v4h*)row0)[lane]       = hs0;   // cols [0,128)  : self
            ((v4h*)(row0 + D))[lane] = ha0;   // cols [128,256): mean(neigh)
            ((v4h*)row1)[lane]       = hs1;
            ((v4h*)(row1 + D))[lane] = ha1;
        }
    }

    __syncthreads();   // sW visible to all waves

    // ---------- Phase 2: WMMA GEMM  C[16x128] = A[16x256] * W[256x128] ----------
    v8f acc[8];
    #pragma unroll
    for (int nt = 0; nt < 8; ++nt) acc[nt] = (v8f){};

    const int m  = lane & 15;
    const int hi = lane >> 4;          // 0 or 1 (lane half)
    const _Float16* arow = &sA[(wv * TILE_M + m) * KPAD];

    #pragma unroll
    for (int ks = 0; ks < 8; ++ks) {
        const int kb = ks * 32;
        // A fragment (16x32 f16): lane m holds K = kb+16*hi+[0,16)
        AFrag ua;
        ua.h[0] = *(const v8h*)(arow + kb + hi * 16);
        ua.h[1] = *(const v8h*)(arow + kb + hi * 16 + 8);

        // B fragments double-buffered so ds_load of nt+1 overlaps WMMA of nt
        AFrag ub[2];
        {
            const _Float16* b0 = &sW[(0 * 16 + m) * KPAD + kb + hi * 16];
            ub[0].h[0] = *(const v8h*)(b0);
            ub[0].h[1] = *(const v8h*)(b0 + 8);
        }
        #pragma unroll
        for (int nt = 0; nt < 8; ++nt) {
            if (nt < 7) {
                const _Float16* bn = &sW[((nt + 1) * 16 + m) * KPAD + kb + hi * 16];
                ub[(nt + 1) & 1].h[0] = *(const v8h*)(bn);
                ub[(nt + 1) & 1].h[1] = *(const v8h*)(bn + 8);
            }
            acc[nt] = __builtin_amdgcn_wmma_f32_16x16x32_f16(
                /*neg_a=*/false, ua.v, /*neg_b=*/false, ub[nt & 1].v,
                /*c_mod=*/(short)0, acc[nt],
                /*reuse_a=*/false, /*reuse_b=*/false);
        }
    }

    // ---------- Store with ReLU (C layout: VGPR r -> M = r + 8*hi, N = m) ----------
    #pragma unroll
    for (int nt = 0; nt < 8; ++nt) {
        const int ncol = nt * 16 + m;
        #pragma unroll
        for (int r = 0; r < 8; ++r) {
            const int row = tileBase + hi * 8 + r;
            if (row < N) {
                __builtin_nontemporal_store(fmaxf(acc[nt][r], 0.0f),
                                            out + (size_t)row * D + ncol);
            }
        }
    }
}

extern "C" void kernel_launch(void* const* d_in, const int* in_sizes, int n_in,
                              void* d_out, int out_size, void* d_ws, size_t ws_size,
                              hipStream_t stream) {
    const float* src     = (const float*)d_in[0];
    const float* src_neg = (const float*)d_in[1];
    const float* dst     = (const float*)d_in[2];
    const float* dst_neg = (const float*)d_in[3];
    const float* w       = (const float*)d_in[4];
    float* out = (float*)d_out;

    const int N = in_sizes[0] / D;                  // 50000
    const int nblocks = (N + NODES_PER_BLOCK - 1) / NODES_PER_BLOCK;

    sage_layer_fused<<<nblocks, BLOCK, 0, stream>>>(src, src_neg, w, out, N);
    sage_layer_fused<<<nblocks, BLOCK, 0, stream>>>(dst, dst_neg, w,
                                                    out + (size_t)N * D, N);
}
